// SparseMHA_30709016166454
// MI455X (gfx1250) — matrix operand order repaired and checked by reading, not compiled, not run on hardware
//
#include <hip/hip_runtime.h>
#include <hip/hip_bf16.h>

typedef __attribute__((ext_vector_type(16))) __bf16 v16bf;
typedef __attribute__((ext_vector_type(8)))  float  v8f;
typedef __attribute__((ext_vector_type(8)))  __bf16 bf16x8;
typedef __attribute__((ext_vector_type(4)))  __bf16 bf16x4;

#define HIDDEN 256
#define HEADS 8
#define HEAD_DIM 32

// ---------------------------------------------------------------------------
// CDNA5 async global->LDS staging (ASYNCcnt-tracked), with sync fallback.
// Builtin signature (from hipcc diagnostics):
//   (v2i/v4i addrspace(1)* gsrc, v2i/v4i addrspace(3)* ldst, Imm off, Imm cpol)
// ---------------------------------------------------------------------------
#if defined(__has_builtin)
#if __has_builtin(__builtin_amdgcn_global_load_async_to_lds_b64) && \
    __has_builtin(__builtin_amdgcn_global_load_async_to_lds_b128) && \
    __has_builtin(__builtin_amdgcn_s_wait_asynccnt)
#define HAS_ASYNC_LDS 1
#endif
#endif
#ifndef HAS_ASYNC_LDS
#define HAS_ASYNC_LDS 0
#endif

#if HAS_ASYNC_LDS
typedef __attribute__((ext_vector_type(2))) int v2i;
typedef __attribute__((ext_vector_type(4))) int v4i;
typedef __attribute__((address_space(1))) v2i* gv2i_p;
typedef __attribute__((address_space(3))) v2i* lv2i_p;
typedef __attribute__((address_space(1))) v4i* gv4i_p;
typedef __attribute__((address_space(3))) v4i* lv4i_p;

__device__ __forceinline__ void async_copy_b64(__bf16* l, const __bf16* g) {
    __builtin_amdgcn_global_load_async_to_lds_b64((gv2i_p)g, (lv2i_p)l, 0, 0);
}
__device__ __forceinline__ void async_copy_b128(__bf16* l, const __bf16* g) {
    __builtin_amdgcn_global_load_async_to_lds_b128((gv4i_p)g, (lv4i_p)l, 0, 0);
}
#endif

// ---------------------------------------------------------------------------
// Elementwise convert h (f32) -> hb (bf16), zero the padded tail rows.
// ---------------------------------------------------------------------------
__global__ __launch_bounds__(256)
void convert_h_kernel(const float* __restrict__ h, __bf16* __restrict__ hb,
                      int n_valid, int n_total)
{
    const int i = blockIdx.x * blockDim.x + threadIdx.x;
    if (i >= n_total) return;
    hb[i] = (i < n_valid) ? (__bf16)h[i] : (__bf16)0.f;
}

// ---------------------------------------------------------------------------
// Convert + transpose the four 256x256 weights to bf16 Wt[n][k] (512 KB,
// L2-resident). One block per weight matrix.
// ---------------------------------------------------------------------------
__global__ __launch_bounds__(256)
void prep_weights_kernel(const float* __restrict__ W0, const float* __restrict__ W1,
                         const float* __restrict__ W2, const float* __restrict__ W3,
                         __bf16* __restrict__ Wt)
{
    const int w = blockIdx.x;
    const float* W = (w == 0) ? W0 : (w == 1) ? W1 : (w == 2) ? W2 : W3;
    __bf16* T = Wt + (size_t)w * HIDDEN * HIDDEN;
    for (int i = threadIdx.x; i < HIDDEN * HIDDEN; i += 256) {
        const int k = i >> 8;     // fan_in
        const int n = i & 255;    // fan_out
        T[(size_t)n * HIDDEN + k] = (__bf16)W[i];
    }
}

// ---------------------------------------------------------------------------
// Tiled GEMM: out[M x 256] = A[M x 256] * W[256 x 256] (+bias)*scale
// A is bf16 (rows padded to 32), Wt is bf16 pre-transposed [n][k].
// bf16 WMMA 16x16x32, f32 accumulate. Block: 256 threads (8 waves),
// block tile 32 x 256, wave tile 16 x 64 (4 accumulators).
// LDS strides of 40 elems (80 B): 16B-aligned rows, conflict-free spread.
// ---------------------------------------------------------------------------
template <bool OUT_BF16>
__global__ __launch_bounds__(256)
void gemm256(const __bf16* __restrict__ A,
             const __bf16* __restrict__ Wt,
             const float* __restrict__ bias,
             float scale, int M,
             __bf16* __restrict__ outb,
             float* __restrict__ outf)
{
    __shared__ __align__(16) __bf16 As[32][40];   // [row][k]
    __shared__ __align__(16) __bf16 Bs[256][40];  // [n][k]

    const int tid  = threadIdx.x;
    const int lane = tid & 31;
    const int wave = tid >> 5;
    const int half = lane >> 4;      // 0: lanes 0-15, 1: lanes 16-31
    const int l16  = lane & 15;
    const int rt   = wave >> 2;      // row tile 0..1
    const int ct0  = (wave & 3) * 4; // first of 4 col tiles
    const int row0 = blockIdx.x * 32;

    v8f acc[4] = {};

    for (int kc = 0; kc < 256; kc += 32) {
        // ---- stage A tile (32 rows x 32 k, bf16): thread t copies 8 B ----
        {
            const int r   = tid >> 3;       // 0..31
            const int seg = (tid & 7) * 4;  // elems 0,4,...,28
            const __bf16* g = A + (size_t)(row0 + r) * HIDDEN + kc + seg;
#if HAS_ASYNC_LDS
            async_copy_b64(&As[r][seg], g);
#else
            *(bf16x4*)&As[r][seg] = *(const bf16x4*)g;
#endif
        }
        // ---- stage B tile (256 n x 32 k, bf16): thread t=n copies 64 B ----
        {
            const int n = tid;
            const __bf16* g = Wt + (size_t)n * HIDDEN + kc;
#if HAS_ASYNC_LDS
            async_copy_b128(&Bs[n][0],  g);
            async_copy_b128(&Bs[n][8],  g + 8);
            async_copy_b128(&Bs[n][16], g + 16);
            async_copy_b128(&Bs[n][24], g + 24);
#else
            *(bf16x8*)&Bs[n][0]  = *(const bf16x8*)(g);
            *(bf16x8*)&Bs[n][8]  = *(const bf16x8*)(g + 8);
            *(bf16x8*)&Bs[n][16] = *(const bf16x8*)(g + 16);
            *(bf16x8*)&Bs[n][24] = *(const bf16x8*)(g + 24);
#endif
        }
#if HAS_ASYNC_LDS
        __builtin_amdgcn_s_wait_asynccnt(0);
#endif
        __syncthreads();

        // ---- A fragment for this wave's row tile (ISA 16-bit A layout) ----
        v16bf a;
        #pragma unroll
        for (int i = 0; i < 16; ++i) {
            const int k = ((i < 8) ? i : (i + 8)) + half * 8;
            a[i] = As[rt * 16 + l16][k];
        }
        // ---- 4 col tiles: B fragment + WMMA ----
        #pragma unroll
        for (int j = 0; j < 4; ++j) {
            v16bf b;
            #pragma unroll
            for (int i = 0; i < 16; ++i) {
                const int k = ((i < 8) ? i : (i + 8)) + half * 8;
                b[i] = Bs[(ct0 + j) * 16 + l16][k];
            }
            acc[j] = __builtin_amdgcn_wmma_f32_16x16x32_bf16(
                false, a, false, b, (short)0, acc[j], false, false);
        }
        __syncthreads();
    }

    // ---- epilogue: (acc + bias) * scale, ISA 32-bit C/D layout ----
    #pragma unroll
    for (int j = 0; j < 4; ++j) {
        const int ccol = (ct0 + j) * 16 + l16;
        const float bc = bias[ccol];
        #pragma unroll
        for (int r = 0; r < 8; ++r) {
            const int grow = row0 + rt * 16 + half * 8 + r;
            if (grow < M) {
                const float v = (acc[j][r] + bc) * scale;
                if (OUT_BF16) outb[(size_t)grow * HIDDEN + ccol] = (__bf16)v;
                else          outf[(size_t)grow * HIDDEN + ccol] = v;
            }
        }
    }
}

// ---------------------------------------------------------------------------
// CSR row_ptr via binary search on sorted `row` (no scan, no atomics).
// ---------------------------------------------------------------------------
__global__ __launch_bounds__(256)
void rowptr_kernel(const int* __restrict__ row, int E, int Nn,
                   int* __restrict__ row_ptr)
{
    const int n = blockIdx.x * blockDim.x + threadIdx.x;
    if (n > Nn) return;
    int lo = 0, hi = E;
    while (lo < hi) {
        const int mid = (lo + hi) >> 1;
        if (row[mid] < n) lo = mid + 1; else hi = mid;
    }
    row_ptr[n] = lo;
}

// ---------------------------------------------------------------------------
// SDDMM: s[e][h] = dot32(q[row[e]][h], k[col[e]][h]); thread per (e, h).
// q/k are bf16 (L2-resident); 64B vectorized reads per operand.
// ---------------------------------------------------------------------------
__global__ __launch_bounds__(256)
void sddmm_kernel(const __bf16* __restrict__ qb, const __bf16* __restrict__ kb,
                  const int* __restrict__ row, const int* __restrict__ col,
                  int E, float* __restrict__ s)
{
    const int t = blockIdx.x * blockDim.x + threadIdx.x;
    if (t >= E * HEADS) return;
    const int e = t >> 3;
    const int h = t & 7;
    const int r = row[e];
    const int c = col[e];
    const __bf16* qp = qb + (size_t)r * HIDDEN + h * HEAD_DIM;
    const __bf16* kp = kb + (size_t)c * HIDDEN + h * HEAD_DIM;
    float acc = 0.f;
    #pragma unroll
    for (int j = 0; j < 4; ++j) {
        const bf16x8 qv = *(const bf16x8*)(qp + 8 * j);
        const bf16x8 kv = *(const bf16x8*)(kp + 8 * j);
        #pragma unroll
        for (int i = 0; i < 8; ++i)
            acc += (float)qv[i] * (float)kv[i];
    }
    s[t] = acc;
}

// ---------------------------------------------------------------------------
// Fused segment softmax + SpMM. One block (256 thr = 8 heads x 32 dims)
// per destination node; one wave32 per head for shuffle reductions.
// ---------------------------------------------------------------------------
__global__ __launch_bounds__(256)
void softmax_spmm_kernel(const float* __restrict__ s,
                         const __bf16* __restrict__ vb,
                         const int* __restrict__ col,
                         const int* __restrict__ row_ptr,
                         __bf16* __restrict__ o)
{
    const int n    = blockIdx.x;
    const int tid  = threadIdx.x;
    const int lane = tid & 31;
    const int h    = tid >> 5;     // head 0..7 (one wave per head)
    const int rs   = row_ptr[n];
    const int re   = row_ptr[n + 1];

    __shared__ float mh[HEADS];
    __shared__ float dh[HEADS];

    // phase 1: per-head max and exp-sum over this node's incoming edges
    float lm = -3.4e38f;
    for (int e = rs + lane; e < re; e += 32)
        lm = fmaxf(lm, s[(size_t)e * HEADS + h]);
    #pragma unroll
    for (int off = 16; off > 0; off >>= 1)
        lm = fmaxf(lm, __shfl_xor(lm, off, 32));

    float ls = 0.f;
    for (int e = rs + lane; e < re; e += 32)
        ls += __expf(s[(size_t)e * HEADS + h] - lm);
    #pragma unroll
    for (int off = 16; off > 0; off >>= 1)
        ls += __shfl_xor(ls, off, 32);

    if (lane == 0) { mh[h] = lm; dh[h] = ls; }
    __syncthreads();

    // phase 2: out[n][h][d] = sum_e softmax(s) * v[col[e]][h][d]
    const float m   = mh[h];
    const float den = dh[h];
    const float inv = (den > 0.f) ? 1.f / den : 0.f;
    float acc = 0.f;
    for (int e = rs; e < re; ++e) {
        const float w = __expf(s[(size_t)e * HEADS + h] - m) * inv;
        const int  c  = col[e];
        acc += w * (float)vb[(size_t)c * HIDDEN + h * HEAD_DIM + lane];
    }
    o[(size_t)n * HIDDEN + tid] = (__bf16)acc;
}

// ---------------------------------------------------------------------------
extern "C" void kernel_launch(void* const* d_in, const int* in_sizes, int n_in,
                              void* d_out, int out_size, void* d_ws, size_t ws_size,
                              hipStream_t stream)
{
    const float* h   = (const float*)d_in[0];
    const int*   row = (const int*)d_in[1];
    const int*   col = (const int*)d_in[2];
    const float* Wq  = (const float*)d_in[3];
    const float* bq  = (const float*)d_in[4];
    const float* Wk  = (const float*)d_in[5];
    const float* bk  = (const float*)d_in[6];
    const float* Wv  = (const float*)d_in[7];
    const float* bv  = (const float*)d_in[8];
    const float* Wo  = (const float*)d_in[9];
    const float* bo  = (const float*)d_in[10];
    float* out = (float*)d_out;

    const int Nn   = in_sizes[0] / HIDDEN;       // 50000
    const int Ee   = in_sizes[1];                // 800000
    const int Npad = (Nn + 31) & ~31;            // padded row count

    // workspace partition (256B aligned)
    char* ws = (char*)d_ws;
    size_t off = 0;
    auto carve = [&](size_t bytes) {
        void* p = ws + off;
        off = (off + bytes + 255) & ~(size_t)255;
        return p;
    };
    const size_t mat_b = (size_t)Npad * HIDDEN * sizeof(__bf16);
    __bf16* hb  = (__bf16*)carve(mat_b);
    __bf16* qb  = (__bf16*)carve(mat_b);
    __bf16* kb  = (__bf16*)carve(mat_b);
    __bf16* vb  = (__bf16*)carve(mat_b);
    __bf16* ob  = (__bf16*)carve(mat_b);
    __bf16* Wt  = (__bf16*)carve((size_t)4 * HIDDEN * HIDDEN * sizeof(__bf16));
    float*  s   = (float*)carve((size_t)Ee * HEADS * sizeof(float));
    int* row_ptr = (int*)carve((size_t)(Nn + 1) * sizeof(int));

    const dim3 blk(256);
    const dim3 gemm_grid(Npad / 32);
    const float scaling = 0.17677669529663687f; // HEAD_DIM^-0.5

    // 0) one-time bf16 conversions: h -> hb (padded), W* -> transposed bf16
    convert_h_kernel<<<((size_t)Npad * HIDDEN + 255) / 256, blk, 0, stream>>>(
        h, hb, Nn * HIDDEN, Npad * HIDDEN);
    prep_weights_kernel<<<4, blk, 0, stream>>>(Wq, Wk, Wv, Wo, Wt);

    // 1) QKV projections (bf16 WMMA, async-LDS staged, bf16 outputs)
    const size_t wsz = (size_t)HIDDEN * HIDDEN;
    gemm256<true><<<gemm_grid, blk, 0, stream>>>(hb, Wt + 0 * wsz, bq, scaling, Nn, qb, nullptr);
    gemm256<true><<<gemm_grid, blk, 0, stream>>>(hb, Wt + 1 * wsz, bk, 1.0f,    Nn, kb, nullptr);
    gemm256<true><<<gemm_grid, blk, 0, stream>>>(hb, Wt + 2 * wsz, bv, 1.0f,    Nn, vb, nullptr);

    // 2) CSR row offsets from sorted row indices
    rowptr_kernel<<<(Nn + 1 + 255) / 256, blk, 0, stream>>>(row, Ee, Nn, row_ptr);

    // 3) per-edge per-head scores
    sddmm_kernel<<<((size_t)Ee * HEADS + 255) / 256, blk, 0, stream>>>(qb, kb, row, col, Ee, s);

    // 4) fused segment softmax + SpMM -> o (bf16)
    softmax_spmm_kernel<<<Nn, blk, 0, stream>>>(s, vb, col, row_ptr, ob);

    // 5) output projection -> d_out (f32)
    gemm256<false><<<gemm_grid, blk, 0, stream>>>(ob, Wt + 3 * wsz, bo, 1.0f, Nn, nullptr, out);
}
